// GTBlock_75505525064439
// MI455X (gfx1250) — matrix operand order repaired and checked by reading, not compiled, not run on hardware
//
#include <hip/hip_runtime.h>
#include <hip/hip_bf16.h>

// ---------------------------------------------------------------------------
// Problem constants (from reference)
// ---------------------------------------------------------------------------
#define Bn   32
#define Tn   512
#define Vn   32000
#define EMBn 128
#define Hn   128
#define H4   512      // 4*H
#define FDn  256
#define GHn  256
#define Nn   50000
#define En   800000
#define Gn   32

typedef __attribute__((ext_vector_type(16))) _Float16 v16h;
typedef __attribute__((ext_vector_type(8)))  float    v8f;
typedef __attribute__((ext_vector_type(4)))  int      v4i;

// Address-space-qualified int4 for the async LDS-DMA builtin:
// arg0 = global (AS1) int4*, arg1 = LDS (AS3) int4*, then imm offset, imm cpol.
typedef __attribute__((address_space(1))) v4i gv4i;   // global int4
typedef __attribute__((address_space(3))) v4i lv4i;   // LDS int4

#if __has_builtin(__builtin_amdgcn_global_load_async_to_lds_b128)
#define HAVE_ASYNC_LDS 1
#else
#define HAVE_ASYNC_LDS 0
#endif

#if __has_builtin(__builtin_amdgcn_s_wait_asynccnt)
#define WAIT_ASYNC() __builtin_amdgcn_s_wait_asynccnt(0)
#else
#define WAIT_ASYNC() ((void)0)
#endif

// ---------------------------------------------------------------------------
// WMMA helpers (wave32, V_WMMA_F32_16X16X32_F16 layouts per CDNA5 ISA 7.12.2)
// A (16x32 f16): lane: m = lane&15, half = lane>>4.
//   elems j=0..7  -> K = k0 + half*8 + j
//   elems j=8..15 -> K = k0 + 16 + half*8 + (j-8)
// B (32x16 f16): lane: n = lane&15, half = lane>>4. same K mapping as A
// C/D (16x16 f32): acc[r] -> row m0 + r + 8*half, col n0 + (lane&15)
// ---------------------------------------------------------------------------
__device__ inline v16h load_a_f16(const _Float16* base, int ldk, int m0, int k0, int lane) {
    int m    = m0 + (lane & 15);
    int half = lane >> 4;
    const _Float16* p = base + (size_t)m * ldk + k0 + half * 8;
    v16h a;
#pragma unroll
    for (int j = 0; j < 8; ++j) a[j] = p[j];
#pragma unroll
    for (int j = 0; j < 8; ++j) a[8 + j] = p[16 + j];
    return a;
}

__device__ inline v16h load_b_f16(const _Float16* W, int ldk, int n0, int k0, int lane) {
    int n    = n0 + (lane & 15);
    int half = lane >> 4;
    const _Float16* p = W + (size_t)n * ldk + k0 + half * 8;
    v16h b;
#pragma unroll
    for (int j = 0; j < 8; ++j) b[j] = p[j];
#pragma unroll
    for (int j = 0; j < 8; ++j) b[8 + j] = p[16 + j];
    return b;
}

__device__ inline v8f wmma_f16(v16h a, v16h b, v8f c) {
    return __builtin_amdgcn_wmma_f32_16x16x32_f16(false, a, false, b, (short)0, c, false, false);
}

__device__ inline float sigm(float x) { return 1.0f / (1.0f + __expf(-x)); }

// ---------------------------------------------------------------------------
// Small utility kernels
// ---------------------------------------------------------------------------
__global__ void cvt_f16_kernel(const float* __restrict__ a, _Float16* __restrict__ o, int n) {
    int i = blockIdx.x * blockDim.x + threadIdx.x;
    if (i < n) o[i] = (_Float16)a[i];
}

__global__ void zero_f32_kernel(float* __restrict__ a, int n) {
    int i = blockIdx.x * blockDim.x + threadIdx.x;
    if (i < n) a[i] = 0.0f;
}

// tok16[row,k] = f16( seq[row] != 0 ? emb[seq[row], k] : 0 )
__global__ void embed_kernel(const int* __restrict__ seq, const float* __restrict__ emb,
                             _Float16* __restrict__ tok16, int total) {
    int idx = blockIdx.x * blockDim.x + threadIdx.x;
    if (idx >= total * EMBn) return;
    int row = idx >> 7;
    int k   = idx & 127;
    int id  = seq[row];
    float v = (id != 0) ? emb[(size_t)id * EMBn + k] : 0.0f;
    tok16[idx] = (_Float16)v;
}

// ---------------------------------------------------------------------------
// Generic WMMA GEMM: out(M,N) = A16(M,K) @ W16(N,K)^T [+ bias][+ addsrc][relu]
// block = 256 threads = 8 waves; all waves share one 16-row A panel, each wave
// owns a 16-col B tile. The A chunk (16x32 f16, 1KB) is staged into LDS via
// GLOBAL_LOAD_ASYNC_TO_LDS_B128 (64 b128 transfers) and shared by all 8 waves
// (8x reduction of A-operand global traffic vs per-wave loads).
// grid = (M/16, N/128). Grids are always exact; no early returns (barriers!).
// ---------------------------------------------------------------------------
__global__ __launch_bounds__(256)
void wmma_gemm_kernel(const _Float16* __restrict__ A, const _Float16* __restrict__ W,
                      const float* __restrict__ bias, const float* __restrict__ addsrc,
                      float* __restrict__ out, int M, int N, int K, int do_relu) {
    __shared__ _Float16 ldsA[16 * 32];   // one K-chunk of the A panel (1 KB)

    int lane = threadIdx.x & 31;
    int wave = threadIdx.x >> 5;
    int m0   = blockIdx.x * 16;
    int n0   = (blockIdx.y * 8 + wave) * 16;

    int col  = n0 + (lane & 15);
    int half = lane >> 4;

    v8f acc;
    float bv = bias ? bias[col] : 0.0f;
#pragma unroll
    for (int r = 0; r < 8; ++r) acc[r] = bv;

    for (int k0 = 0; k0 < K; k0 += 32) {
        // --- stage A[m0..m0+15, k0..k0+31] into LDS ---
        if (threadIdx.x < 64) {
            int row = threadIdx.x >> 2;   // 0..15
            int seg = threadIdx.x & 3;    // 8 f16 (16B) per segment
            const _Float16* src = A + (size_t)(m0 + row) * K + k0 + seg * 8;
            _Float16*       dst = &ldsA[row * 32 + seg * 8];
#if HAVE_ASYNC_LDS
            __builtin_amdgcn_global_load_async_to_lds_b128((gv4i*)src, (lv4i*)dst, 0, 0);
#else
#pragma unroll
            for (int j = 0; j < 8; ++j) dst[j] = src[j];
#endif
        }
#if HAVE_ASYNC_LDS
        WAIT_ASYNC();
#endif
        __syncthreads();

        v16h a = load_a_f16(ldsA, 32, 0, 0, lane);
        v16h b = load_b_f16(W, K, n0, k0, lane);
        acc = wmma_f16(a, b, acc);
        __syncthreads();   // protect ldsA before next chunk overwrites it
    }

#pragma unroll
    for (int r = 0; r < 8; ++r) {
        int row = m0 + r + 8 * half;
        size_t oi = (size_t)row * N + col;
        float v = acc[r];
        if (addsrc) v += addsrc[oi];
        if (do_relu) v = fmaxf(v, 0.0f);
        out[oi] = v;
    }
}

// ---------------------------------------------------------------------------
// Bidirectional LSTM scan. grid = 2 blocks (fwd/bwd), block = 1024 (32 waves).
// Wave w owns gate columns [w*16, w*16+16); W_hh B-tiles pinned in VGPRs.
// LDS: gates f16 (32x512, 32KB), c f32 (32x128, 16KB), h f16 (32x128, 8KB).
// ---------------------------------------------------------------------------
__global__ __launch_bounds__(1024)
void lstm_kernel(const float* __restrict__ xgF, const float* __restrict__ xgB,
                 const _Float16* __restrict__ WhhF16, const _Float16* __restrict__ WhhB16,
                 const float* __restrict__ bhhF, const float* __restrict__ bhhB,
                 const int* __restrict__ lengths,
                 float* __restrict__ hF, float* __restrict__ hB) {
    __shared__ _Float16 gates[Bn * H4];   // 32 KB
    __shared__ float    cs[Bn * Hn];      // 16 KB
    __shared__ _Float16 h16s[Bn * Hn];    //  8 KB

    int dir = blockIdx.x;
    const float*    xg   = dir ? xgB    : xgF;
    const _Float16* Whh  = dir ? WhhB16 : WhhF16;
    const float*    bhh  = dir ? bhhB   : bhhF;
    float*          hout = dir ? hB     : hF;

    int lane = threadIdx.x & 31;
    int wave = threadIdx.x >> 5;      // 0..31 -> n-tile
    int n0   = wave * 16;
    int col  = n0 + (lane & 15);
    int half = lane >> 4;

    // Pin W_hh B-tiles for this wave's 16 output columns across the whole scan.
    v16h breg[4];
#pragma unroll
    for (int kk = 0; kk < 4; ++kk) breg[kk] = load_b_f16(Whh, Hn, n0, kk * 32, lane);

    float bias0 = bhh[col];

    for (int idx = threadIdx.x; idx < Bn * Hn; idx += 1024) {
        cs[idx]   = 0.0f;
        h16s[idx] = (_Float16)0.0f;
    }
    __syncthreads();

    for (int t = 0; t < Tn; ++t) {
        int tt = dir ? (Tn - 1 - t) : t;

        // gates(:, col) = xg(:, tt, col) + b_hh(col) + h @ W_hh^T
#pragma unroll
        for (int mt = 0; mt < 2; ++mt) {
            v8f acc;
#pragma unroll
            for (int r = 0; r < 8; ++r) {
                int b = mt * 16 + r + 8 * half;
                acc[r] = xg[((size_t)(b * Tn + tt)) * H4 + col] + bias0;
            }
#pragma unroll
            for (int kk = 0; kk < 4; ++kk) {
                v16h a = load_a_f16(h16s, Hn, mt * 16, kk * 32, lane);
                acc = wmma_f16(a, breg[kk], acc);
            }
#pragma unroll
            for (int r = 0; r < 8; ++r) {
                int b = mt * 16 + r + 8 * half;
                gates[b * H4 + col] = (_Float16)acc[r];
            }
        }
        __syncthreads();

        // elementwise cell update (4096 elements / 1024 threads)
#pragma unroll
        for (int it = 0; it < 4; ++it) {
            int idx = threadIdx.x + it * 1024;
            int b = idx >> 7;
            int j = idx & 127;
            float ig = (float)gates[b * H4 + j];
            float fg = (float)gates[b * H4 + 128 + j];
            float gg = (float)gates[b * H4 + 256 + j];
            float og = (float)gates[b * H4 + 384 + j];
            float cold = cs[idx];
            float cn = sigm(fg) * cold + sigm(ig) * tanhf(gg);
            float hn = sigm(og) * tanhf(cn);
            if (tt < lengths[b]) {
                cs[idx]   = cn;
                h16s[idx] = (_Float16)hn;
            }
        }
        __syncthreads();
    }

    for (int idx = threadIdx.x; idx < Bn * Hn; idx += 1024)
        hout[idx] = (float)h16s[idx];
}

// hcat16[b, 0:128]=hF[b], hcat16[b, 128:256]=hB[b]
__global__ void hcat_kernel(const float* __restrict__ hF, const float* __restrict__ hB,
                            _Float16* __restrict__ hcat16) {
    int idx = blockIdx.x * blockDim.x + threadIdx.x;
    if (idx >= Bn * 2 * Hn) return;
    int b = idx >> 8;
    int j = idx & 255;
    float v = (j < Hn) ? hF[b * Hn + j] : hB[b * Hn + (j - Hn)];
    hcat16[idx] = (_Float16)v;
}

// ---------------------------------------------------------------------------
// GCN support kernels
// ---------------------------------------------------------------------------
__global__ void deg_init_kernel(float* __restrict__ deg, int n) {
    int i = blockIdx.x * blockDim.x + threadIdx.x;
    if (i < n) deg[i] = 1.0f;           // self loop
}
__global__ void deg_edge_kernel(const int* __restrict__ dst, float* __restrict__ deg, int e) {
    int i = blockIdx.x * blockDim.x + threadIdx.x;
    if (i < e) atomicAdd(&deg[dst[i]], 1.0f);
}
__global__ void dis_kernel(float* __restrict__ deg, int n) {
    int i = blockIdx.x * blockDim.x + threadIdx.x;
    if (i < n) deg[i] = rsqrtf(deg[i]);   // deg >= 1 always
}
// layer-1 scalar reduction: sacc[n] = x[n]*dis[n]^2 + sum_{e:dst=n} x[src]*dis[src]*dis[n]
__global__ void s_init_kernel(const float* __restrict__ x, const float* __restrict__ dis,
                              float* __restrict__ sacc, int n) {
    int i = blockIdx.x * blockDim.x + threadIdx.x;
    if (i < n) sacc[i] = x[i] * dis[i] * dis[i];
}
__global__ void s_edge_kernel(const int* __restrict__ src, const int* __restrict__ dst,
                              const float* __restrict__ x, const float* __restrict__ dis,
                              float* __restrict__ sacc, int e) {
    int i = blockIdx.x * blockDim.x + threadIdx.x;
    if (i >= e) return;
    int s = src[i], d = dst[i];
    atomicAdd(&sacc[d], x[s] * dis[s] * dis[d]);
}
// gx1_16[n,j] = f16( relu(sacc[n]*Wg1[j,0] + bg1[j]) )
__global__ void gx1_kernel(const float* __restrict__ sacc, const float* __restrict__ Wg1,
                           const float* __restrict__ bg1, _Float16* __restrict__ gx1_16) {
    int idx = blockIdx.x * blockDim.x + threadIdx.x;
    if (idx >= Nn * GHn) return;
    int n = idx >> 8;
    int j = idx & 255;
    float v = fmaxf(sacc[n] * Wg1[j] + bg1[j], 0.0f);
    gx1_16[idx] = (_Float16)v;
}
// acc2 self-loop init: acc2[n,:] = h2[n,:]*dis[n]^2
__global__ void acc2_init_kernel(const float* __restrict__ h2, const float* __restrict__ dis,
                                 float* __restrict__ acc2) {
    int idx = blockIdx.x * blockDim.x + threadIdx.x;
    if (idx >= Nn * GHn) return;
    int n = idx >> 8;
    float d = dis[n];
    acc2[idx] = h2[idx] * d * d;
}
// scatter: one wave per edge, each lane handles 8 of 256 columns
__global__ void gcn_scatter_kernel(const int* __restrict__ src, const int* __restrict__ dst,
                                   const float* __restrict__ dis, const float* __restrict__ h2,
                                   float* __restrict__ acc2, int e) {
    int gid  = blockIdx.x * blockDim.x + threadIdx.x;
    int edge = gid >> 5;
    int lane = gid & 31;
    if (edge >= e) return;
    int s = src[edge], d = dst[edge];
    float nrm = dis[s] * dis[d];
    const float* hs = h2 + (size_t)s * GHn;
    float* ad = acc2 + (size_t)d * GHn;
#pragma unroll
    for (int c = 0; c < 8; ++c) {
        int j = lane + c * 32;
        atomicAdd(&ad[j], hs[j] * nrm);
    }
}
// pool: gx2 = relu(acc2 + bg2) accumulated per graph; counts per graph
__global__ void pool_kernel(const float* __restrict__ acc2, const float* __restrict__ bg2,
                            const int* __restrict__ batch, float* __restrict__ pooled,
                            float* __restrict__ counts) {
    int idx = blockIdx.x * blockDim.x + threadIdx.x;
    if (idx >= Nn * GHn) return;
    int n = idx >> 8;
    int j = idx & 255;
    int g = batch[n];
    float v = fmaxf(acc2[idx] + bg2[j], 0.0f);
    atomicAdd(&pooled[g * GHn + j], v);
    if (j == 0) atomicAdd(&counts[g], 1.0f);
}
__global__ void pooled16_kernel(const float* __restrict__ pooled, const float* __restrict__ counts,
                                _Float16* __restrict__ pm16) {
    int idx = blockIdx.x * blockDim.x + threadIdx.x;
    if (idx >= Gn * GHn) return;
    int g = idx >> 8;
    pm16[idx] = (_Float16)(pooled[idx] / fmaxf(counts[g], 1.0f));
}

// ---------------------------------------------------------------------------
// Host-side orchestration
// ---------------------------------------------------------------------------
extern "C" void kernel_launch(void* const* d_in, const int* in_sizes, int n_in,
                              void* d_out, int out_size, void* d_ws, size_t ws_size,
                              hipStream_t stream) {
    // inputs per setup_inputs() order
    const int*   seq_pad = (const int*)  d_in[0];
    const int*   lengths = (const int*)  d_in[1];
    const float* x       = (const float*)d_in[2];
    const int*   e_src   = (const int*)  d_in[3];          // edge_index[0]
    const int*   e_dst   = ((const int*) d_in[3]) + En;    // edge_index[1]
    const int*   batch   = (const int*)  d_in[4];
    const float* emb     = (const float*)d_in[5];
    const float* W_ih_f  = (const float*)d_in[6];
    const float* W_hh_f  = (const float*)d_in[7];
    const float* b_ih_f  = (const float*)d_in[8];
    const float* b_hh_f  = (const float*)d_in[9];
    const float* W_ih_b  = (const float*)d_in[10];
    const float* W_hh_b  = (const float*)d_in[11];
    const float* b_ih_b  = (const float*)d_in[12];
    const float* b_hh_b  = (const float*)d_in[13];
    const float* Wt      = (const float*)d_in[14];
    const float* bt      = (const float*)d_in[15];
    const float* Wg1     = (const float*)d_in[16];
    const float* bg1     = (const float*)d_in[17];
    const float* Wg2     = (const float*)d_in[18];
    const float* bg2     = (const float*)d_in[19];
    const float* Wp      = (const float*)d_in[20];
    const float* bp      = (const float*)d_in[21];
    float* out = (float*)d_out;

    // workspace carve-out
    char*  ws  = (char*)d_ws;
    size_t off = 0;
    auto alloc = [&](size_t bytes) -> void* {
        void* p = ws + off;
        off += (bytes + 255) & ~(size_t)255;
        return p;
    };
    _Float16* tok16   = (_Float16*)alloc((size_t)Bn * Tn * EMBn * 2);
    _Float16* WihF16  = (_Float16*)alloc((size_t)H4 * EMBn * 2);
    _Float16* WihB16  = (_Float16*)alloc((size_t)H4 * EMBn * 2);
    _Float16* WhhF16  = (_Float16*)alloc((size_t)H4 * Hn * 2);
    _Float16* WhhB16  = (_Float16*)alloc((size_t)H4 * Hn * 2);
    _Float16* Wt16    = (_Float16*)alloc((size_t)FDn * 2 * Hn * 2);
    _Float16* Wg216   = (_Float16*)alloc((size_t)GHn * GHn * 2);
    _Float16* Wp16    = (_Float16*)alloc((size_t)FDn * GHn * 2);
    float*    xgF     = (float*)   alloc((size_t)Bn * Tn * H4 * 4);
    float*    xgB     = (float*)   alloc((size_t)Bn * Tn * H4 * 4);
    float*    hF      = (float*)   alloc((size_t)Bn * Hn * 4);
    float*    hB      = (float*)   alloc((size_t)Bn * Hn * 4);
    _Float16* hcat16  = (_Float16*)alloc((size_t)Bn * 2 * Hn * 2);
    float*    txb     = (float*)   alloc((size_t)Bn * FDn * 4);
    float*    dis     = (float*)   alloc((size_t)Nn * 4);
    float*    sacc    = (float*)   alloc((size_t)Nn * 4);
    _Float16* gx116   = (_Float16*)alloc((size_t)Nn * GHn * 2);
    float*    h2      = (float*)   alloc((size_t)Nn * GHn * 4);
    float*    acc2    = (float*)   alloc((size_t)Nn * GHn * 4);
    float*    pooled  = (float*)   alloc((size_t)Gn * GHn * 4);
    float*    counts  = (float*)   alloc((size_t)Gn * 4);
    _Float16* pm16    = (_Float16*)alloc((size_t)Gn * GHn * 2);

    const int TB = 256;
    auto blocks = [](long total, int tb) { return (int)((total + tb - 1) / tb); };

    // --- weight conversions to f16 ---
    cvt_f16_kernel<<<blocks(H4 * EMBn, TB), TB, 0, stream>>>(W_ih_f, WihF16, H4 * EMBn);
    cvt_f16_kernel<<<blocks(H4 * EMBn, TB), TB, 0, stream>>>(W_ih_b, WihB16, H4 * EMBn);
    cvt_f16_kernel<<<blocks(H4 * Hn, TB), TB, 0, stream>>>(W_hh_f, WhhF16, H4 * Hn);
    cvt_f16_kernel<<<blocks(H4 * Hn, TB), TB, 0, stream>>>(W_hh_b, WhhB16, H4 * Hn);
    cvt_f16_kernel<<<blocks(FDn * 2 * Hn, TB), TB, 0, stream>>>(Wt, Wt16, FDn * 2 * Hn);
    cvt_f16_kernel<<<blocks(GHn * GHn, TB), TB, 0, stream>>>(Wg2, Wg216, GHn * GHn);
    cvt_f16_kernel<<<blocks(FDn * GHn, TB), TB, 0, stream>>>(Wp, Wp16, FDn * GHn);

    // --- embedding + input projections (WMMA GEMM, M=16384, N=512, K=128) ---
    embed_kernel<<<blocks((long)Bn * Tn * EMBn, TB), TB, 0, stream>>>(seq_pad, emb, tok16, Bn * Tn);
    {
        dim3 grid(Bn * Tn / 16, H4 / 128);
        wmma_gemm_kernel<<<grid, TB, 0, stream>>>(tok16, WihF16, b_ih_f, nullptr, xgF,
                                                  Bn * Tn, H4, EMBn, 0);
        wmma_gemm_kernel<<<grid, TB, 0, stream>>>(tok16, WihB16, b_ih_b, nullptr, xgB,
                                                  Bn * Tn, H4, EMBn, 0);
    }

    // --- bidirectional LSTM scan (2 blocks; WMMA per step) ---
    lstm_kernel<<<2, 1024, 0, stream>>>(xgF, xgB, WhhF16, WhhB16, b_hh_f, b_hh_b,
                                        lengths, hF, hB);

    // --- tx = concat(hF,hB) @ Wt^T + bt ---
    hcat_kernel<<<blocks(Bn * 2 * Hn, TB), TB, 0, stream>>>(hF, hB, hcat16);
    {
        dim3 grid(Bn / 16, FDn / 128);
        wmma_gemm_kernel<<<grid, TB, 0, stream>>>(hcat16, Wt16, bt, nullptr, txb,
                                                  Bn, FDn, 2 * Hn, 0);
    }

    // --- GCN normalization ---
    deg_init_kernel<<<blocks(Nn, TB), TB, 0, stream>>>(dis, Nn);
    deg_edge_kernel<<<blocks(En, TB), TB, 0, stream>>>(e_dst, dis, En);
    dis_kernel<<<blocks(Nn, TB), TB, 0, stream>>>(dis, Nn);

    // --- GCN layer 1 (algebraically reduced to scalar segment-sum) ---
    s_init_kernel<<<blocks(Nn, TB), TB, 0, stream>>>(x, dis, sacc, Nn);
    s_edge_kernel<<<blocks(En, TB), TB, 0, stream>>>(e_src, e_dst, x, dis, sacc, En);
    gx1_kernel<<<blocks((long)Nn * GHn, TB), TB, 0, stream>>>(sacc, Wg1, bg1, gx116);

    // --- GCN layer 2: h2 = gx1 @ Wg2^T (WMMA, M=50000, N=256, K=256) ---
    {
        dim3 grid(Nn / 16, GHn / 128);
        wmma_gemm_kernel<<<grid, TB, 0, stream>>>(gx116, Wg216, nullptr, nullptr, h2,
                                                  Nn, GHn, GHn, 0);
    }
    acc2_init_kernel<<<blocks((long)Nn * GHn, TB), TB, 0, stream>>>(h2, dis, acc2);
    gcn_scatter_kernel<<<blocks((long)En * 32, TB), TB, 0, stream>>>(e_src, e_dst, dis, h2,
                                                                     acc2, En);

    // --- pooling (mean per graph, fused relu(acc2+bg2)) ---
    zero_f32_kernel<<<blocks(Gn * GHn, TB), TB, 0, stream>>>(pooled, Gn * GHn);
    zero_f32_kernel<<<1, TB, 0, stream>>>(counts, Gn);
    pool_kernel<<<blocks((long)Nn * GHn, TB), TB, 0, stream>>>(acc2, bg2, batch, pooled, counts);
    pooled16_kernel<<<blocks(Gn * GHn, TB), TB, 0, stream>>>(pooled, counts, pm16);

    // --- out = relu(pooled_mean @ Wp^T + bp + tx) ---
    {
        dim3 grid(Gn / 16, FDn / 128);
        wmma_gemm_kernel<<<grid, TB, 0, stream>>>(pm16, Wp16, bp, txb, out,
                                                  Gn, FDn, GHn, 1);
    }
}
